// GraphNet_22711787061577
// MI455X (gfx1250) — matrix operand order repaired and checked by reading, not compiled, not run on hardware
//
#include <hip/hip_runtime.h>
#include <hip/hip_bf16.h>
#include <math.h>

#define HID 128
#define OUTD 3
#define KNN_K 3
#define NEG_SLOPE 0.2f
#define NSEG 8

typedef __attribute__((ext_vector_type(2)))  float  v2f;
typedef __attribute__((ext_vector_type(8)))  float  v8f;
typedef __attribute__((ext_vector_type(8)))  __bf16 v8bf;
typedef __attribute__((ext_vector_type(16))) __bf16 v16bf;

// ---------------------------------------------------------------------------
// f32 WMMA GEMM: C[M,HID] = A[M,K] @ B[K,HID].
// One wave computes a 16x128 row strip (8 accumulator tiles); A fragment reused
// across 8 WMMAs per k-step.  Branchless loads: K guards fold at compile time,
// M guard replaced by clamped row.
// ---------------------------------------------------------------------------
template <int K>
__global__ void gemm_wmma(const float* __restrict__ A, const float* __restrict__ B,
                          float* __restrict__ C, int M) {
    const int lane = threadIdx.x & 31;
    const int half = lane >> 4;
    const int l    = lane & 15;
    const int wave = threadIdx.x >> 5;
    const int mbase = (blockIdx.x * (blockDim.x >> 5) + wave) * 16;

    int rowL = mbase + l;                 // clamped load row (branchless)
    if (rowL > M - 1) rowL = M - 1;
    if (rowL < 0) rowL = 0;

    v8f c[8];
#pragma unroll
    for (int nt = 0; nt < 8; ++nt) c[nt] = (v8f){};

    for (int k = 0; k < K; k += 4) {
        v2f a;
#pragma unroll
        for (int v = 0; v < 2; ++v) {
            int kk = k + v + 2 * half;
            a[v] = ((K & 3) == 0 || kk < K) ? A[(long long)rowL * K + kk] : 0.0f;
        }
#pragma unroll
        for (int nt = 0; nt < 8; ++nt) {
            v2f b;
#pragma unroll
            for (int v = 0; v < 2; ++v) {
                int kk = k + v + 2 * half;
                b[v] = ((K & 3) == 0 || kk < K) ? B[kk * HID + nt * 16 + l] : 0.0f;
            }
            c[nt] = __builtin_amdgcn_wmma_f32_16x16x4_f32(false, a, false, b,
                                                          (short)0, c[nt], false, false);
        }
    }
#pragma unroll
    for (int v = 0; v < 8; ++v) {
        int r = mbase + v + 8 * half;
        if (r < M) {
#pragma unroll
            for (int nt = 0; nt < 8; ++nt)
                C[(long long)r * HID + nt * 16 + l] = c[nt][v];
        }
    }
}

// per-node attention dots: e_src[i] = h[i].a_src, e_dst[i] = h[i].a_dst
__global__ void attn_dots(const float* __restrict__ h, const float* __restrict__ a_s,
                          const float* __restrict__ a_d, float* es, float* ed, int n) {
    int warp = (int)((blockIdx.x * blockDim.x + threadIdx.x) >> 5);
    int lane = threadIdx.x & 31;
    if (warp >= n) return;
    float s = 0.f, d = 0.f;
    for (int f = lane; f < HID; f += 32) {
        float hv = h[warp * HID + f];
        s += hv * a_s[f];
        d += hv * a_d[f];
    }
#pragma unroll
    for (int off = 16; off; off >>= 1) {
        s += __shfl_xor(s, off, 32);
        d += __shfl_xor(d, off, 32);
    }
    if (lane == 0) { es[warp] = s; ed[warp] = d; }
}

__global__ void fill_f32(float* p, float v, int count) {
    int i = blockIdx.x * blockDim.x + threadIdx.x;
    if (i < count) p[i] = v;
}

__device__ inline float atomicMaxF(float* a, float v) {
    if (v >= 0.f) return __int_as_float(atomicMax((int*)a, __float_as_int(v)));
    return __uint_as_float(atomicMin((unsigned int*)a, __float_as_uint(v)));
}

// e = leaky_relu(es[src] + ed[dst]); segment max into m[dst]
__global__ void edge_max(const int* __restrict__ ei, int E, int n,
                         const float* __restrict__ es, const float* __restrict__ ed,
                         float* ev, float* m) {
    int k = blockIdx.x * blockDim.x + threadIdx.x;
    int tot = E + n;
    if (k >= tot) return;
    int s = (k < E) ? ei[k]     : (k - E);
    int d = (k < E) ? ei[E + k] : (k - E);
    float e = es[s] + ed[d];
    e = (e > 0.f) ? e : NEG_SLOPE * e;
    ev[k] = e;
    atomicMaxF(&m[d], e);
}

// ex = exp(e - m[dst]); segment sum into den[dst]
__global__ void edge_exp(const int* __restrict__ ei, int E, int n,
                         float* ev, const float* __restrict__ m, float* den) {
    int k = blockIdx.x * blockDim.x + threadIdx.x;
    int tot = E + n;
    if (k >= tot) return;
    int d = (k < E) ? ei[E + k] : (k - E);
    float ex = expf(ev[k] - m[d]);
    ev[k] = ex;
    atomicAdd(&den[d], ex);
}

// agg[dst] += alpha * h[src]; one lane handles 4 features of one edge
__global__ void edge_agg(const int* __restrict__ ei, int E, int n,
                         const float* __restrict__ ev, const float* __restrict__ den,
                         const float* __restrict__ h, float* agg) {
    long long t = (long long)blockIdx.x * blockDim.x + threadIdx.x;
    int k = (int)(t >> 5);
    int lane = (int)(t & 31);
    int tot = E + n;
    if (k >= tot) return;
    int s = (k < E) ? ei[k]     : (k - E);
    int d = (k < E) ? ei[E + k] : (k - E);
    float dn = den[d];
    dn = dn > 1e-16f ? dn : 1e-16f;
    float alpha = ev[k] / dn;
#pragma unroll
    for (int j = 0; j < 4; ++j) {
        int f = lane * 4 + j;
        atomicAdd(&agg[(long long)d * HID + f], alpha * h[(long long)s * HID + f]);
    }
}

__global__ void bias_relu(float* agg, const float* __restrict__ b, int n) {
    int i = blockIdx.x * blockDim.x + threadIdx.x;
    if (i >= n * HID) return;
    float v = agg[i] + b[i & (HID - 1)];
    agg[i] = v > 0.f ? v : 0.f;
}

// float->bf16 copy + optional row squared-norm
__global__ void to_bf16_norm(const float* __restrict__ h, __bf16* hb, float* rnorm, int n) {
    int i = blockIdx.x * blockDim.x + threadIdx.x;
    if (i >= n) return;
    float acc = 0.f;
    for (int f = 0; f < HID; ++f) {
        float v = h[(long long)i * HID + f];
        hb[(long long)i * HID + f] = (__bf16)v;
        acc += v * v;
    }
    if (rnorm) rnorm[i] = acc;
}

// ---------------------------------------------------------------------------
// Fused KNN: wave = 16 collider rows.  A fragments pinned in 32 VGPRs; B for
// chunk ch+1 is prefetched (ping-pong) while chunk ch computes.  Each lane
// keeps a LANE-LOCAL branchless top-3 over the columns it owns (j%16==lane%16)
// -- no cross-lane traffic in the hot loop.  One 3-round shfl merge at the end.
// ---------------------------------------------------------------------------
__global__ void knn_wmma(const __bf16* __restrict__ hc, const __bf16* __restrict__ hr,
                         const float* __restrict__ rnorm,
                         float* __restrict__ pv, int* __restrict__ pi,
                         int nC, int nR, int nSeg, int chunksPerSeg) {
    int g    = blockIdx.x;
    int seg  = blockIdx.y;
    int lane = threadIdx.x & 31;
    int half = lane >> 4;
    int l    = lane & 15;

    __shared__ __bf16 lA[16 * HID];
    for (int t = threadIdx.x; t < 16 * HID; t += 32) {
        int r = g * 16 + t / HID;
        lA[t] = (r < nC) ? hc[(long long)r * HID + (t % HID)] : (__bf16)0.0f;
    }
    __syncthreads();

    // A 16x32 bf16 layout (contiguous, matches B): half h holds K in [kb+16h..kb+16h+15]
    v16bf aReg[4];
#pragma unroll
    for (int kbi = 0; kbi < 4; ++kbi) {
        int kb = kbi * 32;
        aReg[kbi] = *(const v16bf*)&lA[l * HID + kb + 16 * half];
    }

    // lane-local top-3 (sorted ascending) per accumulator row v
    float tv[8][3];
    int   ti[8][3];
#pragma unroll
    for (int v = 0; v < 8; ++v) {
        tv[v][0] = tv[v][1] = tv[v][2] = __builtin_inff();
        ti[v][0] = ti[v][1] = ti[v][2] = -1;
    }

    int nChunks = nR >> 4;                      // nR multiple of 16
    int c0 = seg * chunksPerSeg;
    int c1 = c0 + chunksPerSeg;
    if (c1 > nChunks) c1 = nChunks;

    v16bf bA[4], bB[4];
    float rnA = 0.f, rnB = 0.f;

    auto loadB = [&](int ch, v16bf* b, float& rn) {
        int jr = ch * 16 + l;
        const __bf16* p = &hr[(long long)jr * HID + 16 * half];
#pragma unroll
        for (int kbi = 0; kbi < 4; ++kbi) b[kbi] = *(const v16bf*)(p + kbi * 32);
        rn = rnorm[jr];
    };

    auto compute = [&](int ch, const v16bf* b, float rn) {
        v8f acc = {};
#pragma unroll
        for (int kbi = 0; kbi < 4; ++kbi)
            acc = __builtin_amdgcn_wmma_f32_16x16x32_bf16(false, aReg[kbi], false, b[kbi],
                                                          (short)0, acc, false, false);
        int idx = ch * 16 + l;
#pragma unroll
        for (int v = 0; v < 8; ++v) {
            float val = rn - 2.0f * acc[v];      // |r|^2 - 2 c.r  (rank metric)
            bool lt0 = val < tv[v][0];
            bool lt1 = val < tv[v][1];
            bool lt2 = val < tv[v][2];
            float t0 = tv[v][0], t1 = tv[v][1];
            int   j0 = ti[v][0], j1 = ti[v][1];
            tv[v][2] = lt1 ? t1 : (lt2 ? val : tv[v][2]);
            ti[v][2] = lt1 ? j1 : (lt2 ? idx : ti[v][2]);
            tv[v][1] = lt0 ? t0 : (lt1 ? val : t1);
            ti[v][1] = lt0 ? j0 : (lt1 ? idx : j1);
            tv[v][0] = lt0 ? val : t0;
            ti[v][0] = lt0 ? idx : j0;
        }
    };

    if (c0 < c1) {
        loadB(c0, bA, rnA);
        int ch = c0;
        while (ch < c1) {
            int pre = (ch + 1 < c1) ? ch + 1 : ch;
            loadB(pre, bB, rnB);                 // prefetch while bA computes
            compute(ch, bA, rnA);
            ++ch;
            if (ch >= c1) break;
            pre = (ch + 1 < c1) ? ch + 1 : ch;
            loadB(pre, bA, rnA);
            compute(ch, bB, rnB);
            ++ch;
        }
    }

    // merge 16 lane-local top-3 lists -> per-row top-3 (3 extraction rounds)
#pragma unroll
    for (int v = 0; v < 8; ++v) {
        int p = 0;
        float outv[3];
        int   outi[3];
#pragma unroll
        for (int t = 0; t < 3; ++t) {
            float cand = (p == 0) ? tv[v][0] : ((p == 1) ? tv[v][1] :
                         ((p == 2) ? tv[v][2] : __builtin_inff()));
            int   cidx = (p == 0) ? ti[v][0] : ((p == 1) ? ti[v][1] :
                         ((p == 2) ? ti[v][2] : -1));
            float bvv = cand; int bii = cidx;
#pragma unroll
            for (int off = 8; off; off >>= 1) {
                float ov = __shfl_xor(bvv, off, 16);
                int   oi = __shfl_xor(bii, off, 16);
                if (ov < bvv || (ov == bvv && oi < bii)) { bvv = ov; bii = oi; }
            }
            outv[t] = bvv; outi[t] = bii;
            if (cand == bvv && cidx == bii) ++p;  // this lane's candidate consumed
        }
        if (l == 0) {
            int m = v + 8 * half;
            long long base = (((long long)g * nSeg + seg) * 16 + m) * 3;
            pv[base + 0] = outv[0]; pv[base + 1] = outv[1]; pv[base + 2] = outv[2];
            pi[base + 0] = outi[0]; pi[base + 1] = outi[1]; pi[base + 2] = outi[2];
        }
    }
}

__global__ void knn_merge(const float* __restrict__ pv, const int* __restrict__ pi,
                          int* __restrict__ nn, int nC, int nSeg) {
    int c = blockIdx.x * blockDim.x + threadIdx.x;
    if (c >= nC) return;
    int g = c >> 4, m = c & 15;
    float b0 = __builtin_inff(), b1 = b0, b2 = b0;
    int i0 = 0, i1 = 0, i2 = 0;
    for (int s = 0; s < nSeg; ++s) {
        long long base = (((long long)g * nSeg + s) * 16 + m) * 3;
        for (int t = 0; t < 3; ++t) {
            float v = pv[base + t];
            int   i = pi[base + t];
            if (v < b0 || (v == b0 && i < i0)) {
                b2 = b1; i2 = i1; b1 = b0; i1 = i0; b0 = v; i0 = i;
            } else if (v < b1 || (v == b1 && i < i1)) {
                b2 = b1; i2 = i1; b1 = v; i1 = i;
            } else if (v < b2 || (v == b2 && i < i2)) {
                b2 = v; i2 = i;
            }
        }
    }
    nn[c * 3 + 0] = i0; nn[c * 3 + 1] = i1; nn[c * 3 + 2] = i2;
}

// scatter_mean numerator/denominator: pooled[r] += h_c[c], cnt[r] += 1
__global__ void pool_scatter(const int* __restrict__ nn, const float* __restrict__ hc,
                             float* pooled, float* cnt, int nC) {
    long long t = (long long)blockIdx.x * blockDim.x + threadIdx.x;
    int e = (int)(t >> 5);
    int lane = (int)(t & 31);
    if (e >= nC * KNN_K) return;
    int c = e / KNN_K;
    int r = nn[e];
    if (r < 0) return;
    if (lane == 0) atomicAdd(&cnt[r], 1.0f);
#pragma unroll
    for (int j = 0; j < 4; ++j) {
        int f = lane * 4 + j;
        atomicAdd(&pooled[(long long)r * HID + f], hc[(long long)c * HID + f]);
    }
}

// decoder: out[i] = [h_r[i], pooled[i]/max(cnt,1)] @ W_dec + b_dec
__global__ void decode(const float* __restrict__ hr, const float* __restrict__ pooled,
                       const float* __restrict__ cnt, const float* __restrict__ Wd,
                       const float* __restrict__ bd, float* out, int nR) {
    int i = blockIdx.x * blockDim.x + threadIdx.x;
    if (i >= nR) return;
    float c = cnt[i];
    float inv = 1.0f / (c > 1.f ? c : 1.f);
    float o0 = bd[0], o1 = bd[1], o2 = bd[2];
    for (int f = 0; f < HID; ++f) {
        float v = hr[(long long)i * HID + f];
        o0 += v * Wd[f * OUTD + 0];
        o1 += v * Wd[f * OUTD + 1];
        o2 += v * Wd[f * OUTD + 2];
    }
    for (int f = 0; f < HID; ++f) {
        float v = pooled[(long long)i * HID + f] * inv;
        o0 += v * Wd[(HID + f) * OUTD + 0];
        o1 += v * Wd[(HID + f) * OUTD + 1];
        o2 += v * Wd[(HID + f) * OUTD + 2];
    }
    out[i * OUTD + 0] = o0;
    out[i * OUTD + 1] = o1;
    out[i * OUTD + 2] = o2;
}

// ---------------------------------------------------------------------------
// Host orchestration
// ---------------------------------------------------------------------------
template <int K>
static void gat_layer(const float* x, const int* ei, int E, int n,
                      const float* W, const float* a_s, const float* a_d, const float* b,
                      float* h, float* agg, float* es, float* ed, float* m, float* den,
                      float* ev, hipStream_t stream) {
    int rowsPerBlock = 128;                       // 8 waves x 16 rows
    gemm_wmma<K><<<(n + rowsPerBlock - 1) / rowsPerBlock, 256, 0, stream>>>(x, W, h, n);
    attn_dots<<<(n * 32 + 255) / 256, 256, 0, stream>>>(h, a_s, a_d, es, ed, n);
    fill_f32<<<(n + 255) / 256, 256, 0, stream>>>(m, -__builtin_inff(), n);
    fill_f32<<<(n + 255) / 256, 256, 0, stream>>>(den, 0.f, n);
    fill_f32<<<(n * HID + 255) / 256, 256, 0, stream>>>(agg, 0.f, n * HID);
    int tot = E + n;
    edge_max<<<(tot + 255) / 256, 256, 0, stream>>>(ei, E, n, es, ed, ev, m);
    edge_exp<<<(tot + 255) / 256, 256, 0, stream>>>(ei, E, n, ev, m, den);
    long long aggT = (long long)tot * 32;
    edge_agg<<<(unsigned)((aggT + 255) / 256), 256, 0, stream>>>(ei, E, n, ev, den, h, agg);
    bias_relu<<<(n * HID + 255) / 256, 256, 0, stream>>>(agg, b, n);
}

extern "C" void kernel_launch(void* const* d_in, const int* in_sizes, int n_in,
                              void* d_out, int out_size, void* d_ws, size_t ws_size,
                              hipStream_t stream) {
    const float* x_r  = (const float*)d_in[0];
    const float* x_c  = (const float*)d_in[1];
    const int*   ei_r = (const int*)d_in[2];
    const int*   ei_c = (const int*)d_in[3];
    const float *W_r1 = (const float*)d_in[5],  *as_r1 = (const float*)d_in[6],
                *ad_r1 = (const float*)d_in[7], *b_r1 = (const float*)d_in[8];
    const float *W_r2 = (const float*)d_in[9],  *as_r2 = (const float*)d_in[10],
                *ad_r2 = (const float*)d_in[11], *b_r2 = (const float*)d_in[12];
    const float *W_c1 = (const float*)d_in[13], *as_c1 = (const float*)d_in[14],
                *ad_c1 = (const float*)d_in[15], *b_c1 = (const float*)d_in[16];
    const float *W_c2 = (const float*)d_in[17], *as_c2 = (const float*)d_in[18],
                *ad_c2 = (const float*)d_in[19], *b_c2 = (const float*)d_in[20];
    const float *W_dec = (const float*)d_in[21], *b_dec = (const float*)d_in[22];

    const int D_IN = 6;
    int nR = in_sizes[0] / D_IN;
    int nC = in_sizes[1] / D_IN;
    int E_R = in_sizes[2] / 2;
    int E_C = in_sizes[3] / 2;

    // workspace bump allocator (256B aligned)
    char* ws = (char*)d_ws;
    size_t off = 0;
    auto alloc = [&](size_t bytes) -> char* {
        char* p = ws + off;
        off += (bytes + 255) & ~(size_t)255;
        return p;
    };
    float* hA_r = (float*)alloc((size_t)nR * HID * 4);
    float* hB_r = (float*)alloc((size_t)nR * HID * 4);
    float* hA_c = (float*)alloc((size_t)nC * HID * 4);
    float* hB_c = (float*)alloc((size_t)nC * HID * 4);
    float* es_r = (float*)alloc((size_t)nR * 4);
    float* ed_r = (float*)alloc((size_t)nR * 4);
    float* m_r  = (float*)alloc((size_t)nR * 4);
    float* dn_r = (float*)alloc((size_t)nR * 4);
    float* es_c = (float*)alloc((size_t)nC * 4);
    float* ed_c = (float*)alloc((size_t)nC * 4);
    float* m_c  = (float*)alloc((size_t)nC * 4);
    float* dn_c = (float*)alloc((size_t)nC * 4);
    float* ev_r = (float*)alloc((size_t)(E_R + nR) * 4);
    float* ev_c = (float*)alloc((size_t)(E_C + nC) * 4);
    __bf16* hr_bf = (__bf16*)alloc((size_t)nR * HID * 2);
    __bf16* hc_bf = (__bf16*)alloc((size_t)nC * HID * 2);
    float* rnorm = (float*)alloc((size_t)nR * 4);
    int nGroups = (nC + 15) / 16;
    float* pv = (float*)alloc((size_t)nGroups * NSEG * 16 * 3 * 4);
    int*   pi = (int*)  alloc((size_t)nGroups * NSEG * 16 * 3 * 4);
    int*   nn = (int*)  alloc((size_t)nC * KNN_K * 4);
    float* pooled = (float*)alloc((size_t)nR * HID * 4);
    float* cnt    = (float*)alloc((size_t)nR * 4);
    (void)ws_size;

    // resting branch: layer1 (x_r -> hA, agg -> hB), layer2 (hB -> hA, agg -> hB)
    gat_layer<6>(x_r, ei_r, E_R, nR, W_r1, as_r1, ad_r1, b_r1,
                 hA_r, hB_r, es_r, ed_r, m_r, dn_r, ev_r, stream);
    gat_layer<HID>(hB_r, ei_r, E_R, nR, W_r2, as_r2, ad_r2, b_r2,
                   hA_r, hB_r, es_r, ed_r, m_r, dn_r, ev_r, stream);
    // collider branch
    gat_layer<6>(x_c, ei_c, E_C, nC, W_c1, as_c1, ad_c1, b_c1,
                 hA_c, hB_c, es_c, ed_c, m_c, dn_c, ev_c, stream);
    gat_layer<HID>(hB_c, ei_c, E_C, nC, W_c2, as_c2, ad_c2, b_c2,
                   hA_c, hB_c, es_c, ed_c, m_c, dn_c, ev_c, stream);

    // KNN (bf16 WMMA) + scatter-mean + decoder
    to_bf16_norm<<<(nR + 255) / 256, 256, 0, stream>>>(hB_r, hr_bf, rnorm, nR);
    to_bf16_norm<<<(nC + 255) / 256, 256, 0, stream>>>(hB_c, hc_bf, nullptr, nC);
    int nChunks = nR / 16;
    int chunksPerSeg = (nChunks + NSEG - 1) / NSEG;
    knn_wmma<<<dim3(nGroups, NSEG), 32, 0, stream>>>(hc_bf, hr_bf, rnorm, pv, pi,
                                                     nC, nR, NSEG, chunksPerSeg);
    knn_merge<<<(nC + 255) / 256, 256, 0, stream>>>(pv, pi, nn, nC, NSEG);
    fill_f32<<<(nR * HID + 255) / 256, 256, 0, stream>>>(pooled, 0.f, nR * HID);
    fill_f32<<<(nR + 255) / 256, 256, 0, stream>>>(cnt, 0.f, nR);
    long long st = (long long)nC * KNN_K * 32;
    pool_scatter<<<(unsigned)((st + 255) / 256), 256, 0, stream>>>(nn, hB_c, pooled, cnt, nC);
    decode<<<(nR + 255) / 256, 256, 0, stream>>>(hB_r, pooled, cnt, W_dec, b_dec,
                                                 (float*)d_out, nR);
}
